// MultiHeadAttention_82660940579038
// MI455X (gfx1250) — matrix operand order repaired and checked by reading, not compiled, not run on hardware
//
#include <hip/hip_runtime.h>
#include <hip/hip_bf16.h>

// ---------------------------------------------------------------------------
// MI455X / gfx1250 multi-head attention with RoPE, all matmuls on
// v_wmma_f32_16x16x32_f16 (f16 operands, f32 accumulation).
// B=2, S=2048, HID=1024, NH=16, HD=64.
//
// Pipeline:
//   1) cvt_f32_to_f16  : one streaming pass producing f16 copies of
//                        q,k,v and Wq,Wk,Wv,Wo (removes all conversions
//                        from GEMM inner loops, halves L2 fetch bytes)
//   2) proj_gemm x3    : Q/K head-major f16, V transposed f16
//   3) rope_kernel     : in-place RoPE on Q,K (f32 math)
//   4) attn_kernel     : causal flash attention, online softmax,
//                        per-wave LDS D->A relayout (s_wait_dscnt)
//   5) out_gemm        : head-major AO x Wo^T -> f32 output
// ---------------------------------------------------------------------------

typedef _Float16 half_t;
typedef __attribute__((ext_vector_type(16))) _Float16 v16h;
typedef __attribute__((ext_vector_type(8)))  _Float16 v8h;
typedef __attribute__((ext_vector_type(4)))  _Float16 v4h;
typedef __attribute__((ext_vector_type(8)))  float    v8f;
typedef __attribute__((ext_vector_type(4)))  float    v4f;

#define BB   2
#define SS   2048
#define HIDN 1024
#define NHN  16
#define HDN  64

__device__ __forceinline__ v8f wmma32(v16h a, v16h b, v8f c) {
  return __builtin_amdgcn_wmma_f32_16x16x32_f16(
      /*neg_a=*/false, a, /*neg_b=*/false, b,
      /*c_mod=*/(short)0, c, /*reuse_a=*/false, /*reuse_b=*/false);
}

// A fragment (16x32 f16, M x K): lane holds row m = lane&15, 16 contiguous
// K values [2*kboff, 2*kboff+16) in elements 0..15, where kboff = (lane>>4)*8.
__device__ __forceinline__ v16h load_a_f16(const half_t* rowp, int kboff) {
  v8h lo = *(const v8h*)(rowp + 2 * kboff);
  v8h hi = *(const v8h*)(rowp + 2 * kboff + 8);
  v16h r;
#pragma unroll
  for (int i = 0; i < 8; ++i) { r[i] = lo[i]; r[i + 8] = hi[i]; }
  return r;
}

// B fragment (32x16 f16, K x N): lane holds column n = lane&15, 16 contiguous
// K values starting at k0 = (lane>>4)*16.  p points at 16 contiguous halves.
__device__ __forceinline__ v16h load_b_f16(const half_t* p) {
  v8h lo = *(const v8h*)(p);
  v8h hi = *(const v8h*)(p + 8);
  v16h r;
#pragma unroll
  for (int i = 0; i < 8; ++i) { r[i] = lo[i]; r[i + 8] = hi[i]; }
  return r;
}

// ---------------------------------------------------------------------------
// Kernel 0: streaming f32 -> f16 conversion (4 elems / thread).
// ---------------------------------------------------------------------------
__global__ __launch_bounds__(256) void cvt_f32_to_f16(
    const float* __restrict__ src, half_t* __restrict__ dst, int n4)
{
  const int i = blockIdx.x * 256 + threadIdx.x;
  if (i < n4) {
    v4f x = ((const v4f*)src)[i];
    v4h y;
#pragma unroll
    for (int j = 0; j < 4; ++j) y[j] = (half_t)x[j];
    ((v4h*)dst)[i] = y;
  }
}

// ---------------------------------------------------------------------------
// Kernel 1: projection GEMM  out[n][c] = sum_k X[n][k] * W[c][k]   (all f16)
//   mode 0: write head-major f16  out[b][h][s][d]      (Q, K)
//   mode 1: write transposed f16  out[b][h][d][s]      (V)
// Block: 256 threads (8 waves). Block tile 128 rows x 128 cols.
// Wave tile 32 rows x 64 cols = 2x4 WMMA accumulators; pure load+WMMA loop.
// ---------------------------------------------------------------------------
__global__ __launch_bounds__(256) void proj_gemm(
    const half_t* __restrict__ X, const half_t* __restrict__ W,
    half_t* __restrict__ out, int mode)
{
  const int lane = threadIdx.x & 31;
  const int w    = threadIdx.x >> 5;
  const int wr   = w & 3;          // 0..3 row group
  const int wc   = w >> 2;         // 0..1 col group
  const int row0 = blockIdx.y * 128 + wr * 32;
  const int col0 = blockIdx.x * 128 + wc * 64;
  const int m     = lane & 15;
  const int hgrp  = lane >> 4;
  const int kboff = hgrp * 8;
  const int k0    = hgrp * 16;
  const int m0    = hgrp * 8;
  const int n     = lane & 15;

  v8f acc[2][4];
#pragma unroll
  for (int ri = 0; ri < 2; ++ri)
#pragma unroll
    for (int ci = 0; ci < 4; ++ci)
#pragma unroll
      for (int r = 0; r < 8; ++r) acc[ri][ci][r] = 0.0f;

  for (int kk = 0; kk < HIDN; kk += 32) {
    if (kk + 32 < HIDN) {   // prefetch next k-step (global_prefetch_b8)
      __builtin_prefetch(X + (row0 + m) * HIDN + kk + 32 + kboff, 0, 1);
      __builtin_prefetch(W + (col0 + n) * HIDN + kk + 32 + k0, 0, 1);
    }
    v16h aF[2], bF[4];
#pragma unroll
    for (int ri = 0; ri < 2; ++ri)
      aF[ri] = load_a_f16(X + (row0 + ri * 16 + m) * HIDN + kk, kboff);
#pragma unroll
    for (int ci = 0; ci < 4; ++ci)
      bF[ci] = load_b_f16(W + (col0 + ci * 16 + n) * HIDN + kk + k0);
#pragma unroll
    for (int ri = 0; ri < 2; ++ri)
#pragma unroll
      for (int ci = 0; ci < 4; ++ci)
        acc[ri][ci] = wmma32(aF[ri], bF[ci], acc[ri][ci]);
  }

  // Epilogue: D fragment lane holds column (col0+ci*16+n), rows m0..m0+7.
#pragma unroll
  for (int ri = 0; ri < 2; ++ri) {
    const int rt0 = row0 + ri * 16 + m0;          // first token row (mult of 8)
    const int b   = rt0 >> 11;                    // / 2048
    const int s0  = rt0 & (SS - 1);
#pragma unroll
    for (int ci = 0; ci < 4; ++ci) {
      const int feat = col0 + ci * 16 + n;
      const int h = feat >> 6;
      const int d = feat & 63;
      if (mode == 0) {
        half_t* base = out + (((b * NHN + h) * SS + s0) * HDN + d);
#pragma unroll
        for (int r = 0; r < 8; ++r)
          base[r * HDN] = (half_t)acc[ri][ci][r];
      } else {
        half_t* dst = out + (((b * NHN + h) * HDN + d) * SS + s0);
        v8h t;
#pragma unroll
        for (int r = 0; r < 8; ++r) t[r] = (half_t)acc[ri][ci][r];
        *(v8h*)dst = t;
      }
    }
  }
}

// ---------------------------------------------------------------------------
// Kernel 2: RoPE in-place on Qh and Kh (head-major f16, f32 math).
// One thread per (b,h,s,pair).  2,097,152 threads.
// ---------------------------------------------------------------------------
__global__ __launch_bounds__(256) void rope_kernel(
    half_t* __restrict__ Qh, half_t* __restrict__ Kh)
{
  const int idx = blockIdx.x * 256 + threadIdx.x;
  const int d2 = idx & 31;                 // pair index 0..31
  const int s  = (idx >> 5) & (SS - 1);
  const int hh = idx >> 16;                // b*NH + h, 0..31
  // theta = 10000^(-2*d2/64) = exp(-d2 * ln(1e4)/32)
  const float theta = __expf((float)d2 * -0.28782313662425575f);
  const float ang = (float)s * theta;
  float sn, cs;
  __sincosf(ang, &sn, &cs);
  const int base = (hh * SS + s) * HDN + 2 * d2;

  float qe = (float)Qh[base], qo = (float)Qh[base + 1];
  Qh[base]     = (half_t)(qe * cs - qo * sn);
  Qh[base + 1] = (half_t)(qo * cs + qe * sn);

  float ke = (float)Kh[base], ko = (float)Kh[base + 1];
  Kh[base]     = (half_t)(ke * cs - ko * sn);
  Kh[base + 1] = (half_t)(ko * cs + ke * sn);
}

// ---------------------------------------------------------------------------
// Kernel 3: causal flash attention.
// Grid: (S/64, NH, B). Block: 128 threads = 4 independent waves.
// Each wave owns 16 query rows; streams 32-key blocks with online softmax.
// No cross-wave barriers (waves have different causal trip counts); the
// P (probabilities) D->A relayout goes through a per-wave private LDS strip.
// ---------------------------------------------------------------------------
__global__ __launch_bounds__(128) void attn_kernel(
    const half_t* __restrict__ Qh, const half_t* __restrict__ Kh,
    const half_t* __restrict__ Vt, half_t* __restrict__ Oh)
{
  __shared__ half_t Pbuf[4][16 * 32];      // 4 KiB: per-wave 16x32 P tile

  const int lane = threadIdx.x & 31;
  const int w    = threadIdx.x >> 5;
  const int hh   = blockIdx.z * NHN + blockIdx.y;   // b*NH + h
  const int qrow0 = blockIdx.x * 64 + w * 16;

  const int m     = lane & 15;
  const int hgrp  = lane >> 4;
  const int kboff = hgrp * 8;
  const int k0    = hgrp * 16;
  const int m0    = hgrp * 8;
  const int n     = lane & 15;

  const half_t* Qb = Qh + (size_t)hh * SS * HDN;
  const half_t* Kb = Kh + (size_t)hh * SS * HDN;
  const half_t* Vb = Vt + (size_t)hh * HDN * SS;
  half_t*       Ob = Oh + (size_t)hh * SS * HDN;

  // Q fragments for the whole strip (16 rows x 64 features = 2 K-steps).
  v16h aQ[2];
#pragma unroll
  for (int ks = 0; ks < 2; ++ks)
    aQ[ks] = load_a_f16(Qb + (qrow0 + m) * HDN + ks * 32, kboff);

  v8f O[4];
#pragma unroll
  for (int t = 0; t < 4; ++t)
#pragma unroll
    for (int r = 0; r < 8; ++r) O[t][r] = 0.0f;

  float mi[8], li[8];
#pragma unroll
  for (int r = 0; r < 8; ++r) { mi[r] = -3.0e38f; li[r] = 0.0f; }

  const int nkb = (qrow0 >> 5) + 1;        // causal: keys <= qrow0+15
  for (int kb = 0; kb < nkb; ++kb) {
    const int kj0 = kb * 32;
    const bool lastb = (kb == nkb - 1);

    // --- scores: 16 queries x 32 keys, 4 WMMAs --------------------------
    v8f sc0, sc1;
#pragma unroll
    for (int r = 0; r < 8; ++r) { sc0[r] = 0.0f; sc1[r] = 0.0f; }
#pragma unroll
    for (int ks = 0; ks < 2; ++ks) {
      v16h bK0 = load_b_f16(Kb + (kj0 + n) * HDN + ks * 32 + k0);
      v16h bK1 = load_b_f16(Kb + (kj0 + 16 + n) * HDN + ks * 32 + k0);
      sc0 = wmma32(aQ[ks], bK0, sc0);
      sc1 = wmma32(aQ[ks], bK1, sc1);
    }

    // --- scale + causal mask + online softmax ---------------------------
    float p0[8], p1[8], smax[8];
#pragma unroll
    for (int r = 0; r < 8; ++r) {
      const int qi = qrow0 + m0 + r;
      float s0v = sc0[r] * 0.125f;          // 1/sqrt(64)
      float s1v = sc1[r] * 0.125f;
      if (lastb) {
        if (kj0 + n > qi)      s0v = -3.0e38f;
        if (kj0 + 16 + n > qi) s1v = -3.0e38f;
      }
      p0[r] = s0v; p1[r] = s1v;
      smax[r] = fmaxf(s0v, s1v);
    }
#pragma unroll
    for (int off = 8; off >= 1; off >>= 1)
#pragma unroll
      for (int r = 0; r < 8; ++r)
        smax[r] = fmaxf(smax[r], __shfl_xor(smax[r], off, 32));

    float al[8], rs[8];
#pragma unroll
    for (int r = 0; r < 8; ++r) {
      const float nm = fmaxf(mi[r], smax[r]);
      al[r] = __expf(mi[r] - nm);
      mi[r] = nm;
      p0[r] = __expf(p0[r] - nm);
      p1[r] = __expf(p1[r] - nm);
      rs[r] = p0[r] + p1[r];
    }
#pragma unroll
    for (int off = 8; off >= 1; off >>= 1)
#pragma unroll
      for (int r = 0; r < 8; ++r)
        rs[r] += __shfl_xor(rs[r], off, 32);
#pragma unroll
    for (int r = 0; r < 8; ++r) li[r] = li[r] * al[r] + rs[r];

    // rescale running output
#pragma unroll
    for (int t = 0; t < 4; ++t)
#pragma unroll
      for (int r = 0; r < 8; ++r) O[t][r] *= al[r];

    // --- D-layout -> A-layout relayout of P via per-wave LDS ------------
    half_t* pw = &Pbuf[w][0];
    asm volatile("" ::: "memory");
#pragma unroll
    for (int r = 0; r < 8; ++r) {
      pw[(m0 + r) * 32 + n]      = (half_t)p0[r];
      pw[(m0 + r) * 32 + 16 + n] = (half_t)p1[r];
    }
    asm volatile("s_wait_dscnt 0" ::: "memory");  // wave-local store->load order

    v16h aP = load_a_f16(pw + m * 32, kboff);     // K = this 32-key block

    // --- P @ V : 4 WMMAs (V stored transposed -> contiguous B frags) ----
#pragma unroll
    for (int t = 0; t < 4; ++t) {
      v16h bV = load_b_f16(Vb + (t * 16 + n) * SS + kj0 + k0);
      O[t] = wmma32(aP, bV, O[t]);
    }
  }

  // --- normalize and store head-major ----------------------------------
  float invl[8];
#pragma unroll
  for (int r = 0; r < 8; ++r) invl[r] = 1.0f / li[r];
#pragma unroll
  for (int t = 0; t < 4; ++t) {
    half_t* base = Ob + (qrow0 + m0) * HDN + t * 16 + n;
#pragma unroll
    for (int r = 0; r < 8; ++r)
      base[r * HDN] = (half_t)(O[t][r] * invl[r]);
  }
}

// ---------------------------------------------------------------------------
// Kernel 4: output projection  out[n][c] = sum_k AO[n][k] * Wo16[c][k]
//   AO read from head-major f16 Oh[b][h][s][d] (k = h*64+d), Wo16 f16,
//   result f32 to d_out.  Same tiling as proj_gemm.
// ---------------------------------------------------------------------------
__global__ __launch_bounds__(256) void out_gemm(
    const half_t* __restrict__ Oh, const half_t* __restrict__ Wo,
    float* __restrict__ out)
{
  const int lane = threadIdx.x & 31;
  const int w    = threadIdx.x >> 5;
  const int wr   = w & 3;
  const int wc   = w >> 2;
  const int row0 = blockIdx.y * 128 + wr * 32;
  const int col0 = blockIdx.x * 128 + wc * 64;
  const int m     = lane & 15;
  const int hgrp  = lane >> 4;
  const int kboff = hgrp * 8;
  const int k0    = hgrp * 16;
  const int m0    = hgrp * 8;
  const int n     = lane & 15;

  v8f acc[2][4];
#pragma unroll
  for (int ri = 0; ri < 2; ++ri)
#pragma unroll
    for (int ci = 0; ci < 4; ++ci)
#pragma unroll
      for (int r = 0; r < 8; ++r) acc[ri][ci][r] = 0.0f;

  // Per-row base pointers into head-major AO.
  const half_t* abase[2];
#pragma unroll
  for (int ri = 0; ri < 2; ++ri) {
    const int rt = row0 + ri * 16 + m;
    const int b  = rt >> 11;
    const int s  = rt & (SS - 1);
    abase[ri] = Oh + ((size_t)b * NHN * SS + s) * HDN;   // + h*S*HD + d
  }

  for (int kk = 0; kk < HIDN; kk += 32) {
    if (kk + 32 < HIDN)
      __builtin_prefetch(Wo + (col0 + n) * HIDN + kk + 32 + k0, 0, 1);
    v16h aF[2], bF[4];
#pragma unroll
    for (int ri = 0; ri < 2; ++ri) {
      const int ko0 = kk + 2 * kboff;        // multiple of 8, inside one head
      const int ko1 = kk + 2 * kboff + 8;
      const half_t* c0 = abase[ri] + (size_t)(ko0 >> 6) * SS * HDN + (ko0 & 63);
      const half_t* c1 = abase[ri] + (size_t)(ko1 >> 6) * SS * HDN + (ko1 & 63);
      v8h lo = *(const v8h*)c0;
      v8h hi = *(const v8h*)c1;
      v16h a;
#pragma unroll
      for (int i = 0; i < 8; ++i) { a[i] = lo[i]; a[i + 8] = hi[i]; }
      aF[ri] = a;
    }
#pragma unroll
    for (int ci = 0; ci < 4; ++ci)
      bF[ci] = load_b_f16(Wo + (col0 + ci * 16 + n) * HIDN + kk + k0);
#pragma unroll
    for (int ri = 0; ri < 2; ++ri)
#pragma unroll
      for (int ci = 0; ci < 4; ++ci)
        acc[ri][ci] = wmma32(aF[ri], bF[ci], acc[ri][ci]);
  }

#pragma unroll
  for (int ri = 0; ri < 2; ++ri) {
    const int rt0 = row0 + ri * 16 + m0;
#pragma unroll
    for (int ci = 0; ci < 4; ++ci) {
      const int feat = col0 + ci * 16 + n;
      float* dst = out + (size_t)rt0 * HIDN + feat;
#pragma unroll
      for (int r = 0; r < 8; ++r)
        dst[(size_t)r * HIDN] = acc[ri][ci][r];
    }
  }
}

// ---------------------------------------------------------------------------
// Host-side launcher.
// Inputs (setup_inputs order): q, k, v, mask, Wq, Wk, Wv, Wo.
// Workspace layout (f16):
//   qf,kf,vf   : 3 x 4,194,304 halves  (f16 copies of activations)
//   wq..wo     : 4 x 1,048,576 halves  (f16 copies of weights)
//   Qh,Kh,Vt,Oh: 4 x 4,194,304 halves  (projected / attention buffers)
// Total: 64 MiB.
// ---------------------------------------------------------------------------
extern "C" void kernel_launch(void* const* d_in, const int* in_sizes, int n_in,
                              void* d_out, int out_size, void* d_ws, size_t ws_size,
                              hipStream_t stream) {
  const float* q  = (const float*)d_in[0];
  const float* k  = (const float*)d_in[1];
  const float* v  = (const float*)d_in[2];
  // d_in[3] = mask (causal; implemented analytically in attn_kernel)
  const float* Wq = (const float*)d_in[4];
  const float* Wk = (const float*)d_in[5];
  const float* Wv = (const float*)d_in[6];
  const float* Wo = (const float*)d_in[7];

  const size_t AELEMS = (size_t)BB * SS * HIDN;        // 4,194,304
  const size_t WELEMS = (size_t)HIDN * HIDN;           // 1,048,576

  half_t* qf = (half_t*)d_ws;
  half_t* kf = qf + AELEMS;
  half_t* vf = kf + AELEMS;
  half_t* wq = vf + AELEMS;
  half_t* wk = wq + WELEMS;
  half_t* wv = wk + WELEMS;
  half_t* wo = wv + WELEMS;
  half_t* Qh = wo + WELEMS;
  half_t* Kh = Qh + AELEMS;
  half_t* Vt = Kh + AELEMS;
  half_t* Oh = Vt + AELEMS;

  const int an4 = (int)(AELEMS / 4), wn4 = (int)(WELEMS / 4);
  cvt_f32_to_f16<<<(an4 + 255) / 256, 256, 0, stream>>>(q,  qf, an4);
  cvt_f32_to_f16<<<(an4 + 255) / 256, 256, 0, stream>>>(k,  kf, an4);
  cvt_f32_to_f16<<<(an4 + 255) / 256, 256, 0, stream>>>(v,  vf, an4);
  cvt_f32_to_f16<<<(wn4 + 255) / 256, 256, 0, stream>>>(Wq, wq, wn4);
  cvt_f32_to_f16<<<(wn4 + 255) / 256, 256, 0, stream>>>(Wk, wk, wn4);
  cvt_f32_to_f16<<<(wn4 + 255) / 256, 256, 0, stream>>>(Wv, wv, wn4);
  cvt_f32_to_f16<<<(wn4 + 255) / 256, 256, 0, stream>>>(Wo, wo, wn4);

  dim3 ggemm(HIDN / 128, (BB * SS) / 128);   // (8, 32)
  proj_gemm<<<ggemm, 256, 0, stream>>>(qf, wq, Qh, 0);
  proj_gemm<<<ggemm, 256, 0, stream>>>(kf, wk, Kh, 0);
  proj_gemm<<<ggemm, 256, 0, stream>>>(vf, wv, Vt, 1);

  rope_kernel<<<(BB * NHN * SS * (HDN / 2)) / 256, 256, 0, stream>>>(Qh, Kh);

  attn_kernel<<<dim3(SS / 64, NHN, BB), 128, 0, stream>>>(Qh, Kh, Vt, Oh);

  out_gemm<<<ggemm, 256, 0, stream>>>(Oh, wo, (float*)d_out);
}
